// RNN_Unit_28784870818476
// MI455X (gfx1250) — matrix operand order repaired and checked by reading, not compiled, not run on hardware
//
#include <hip/hip_runtime.h>

#define BSZ 2048
#define HDIM 1024
#define XD 2304           // input_x width
#define INDIM 4352        // XD + 2*HDIM
#define BM 128            // block M tile
#define BH 32             // block H tile (per gate)
#define KT 32             // K step (fp16 WMMA K)
#define NK (INDIM / KT)   // 136 (even)
#define LDSS 40           // padded LDS stride (halves) per 32-K row

#define ZCH ((BSZ * INDIM) / 8)        // fp16-convert chunks for z
#define WCH ((4 * HDIM * INDIM) / 8)   // fp16-convert chunks for W_all

typedef __attribute__((ext_vector_type(16))) _Float16 v16h;
typedef __attribute__((ext_vector_type(8)))  _Float16 v8h;
typedef __attribute__((ext_vector_type(8)))  float    v8f;

union V16 { v16h v; v8h h[2]; };

__device__ __forceinline__ float fast_sig(float x) {
  x = fminf(30.f, fmaxf(-30.f, x));
  return 1.f / (1.f + __expf(-x));
}
__device__ __forceinline__ float fast_tanh(float x) {
  x = fminf(15.f, fmaxf(-15.f, x));
  float e = __expf(2.f * x);
  return (e - 1.f) / (e + 1.f);
}

// Per-lane 16-byte global -> LDS asynchronous copy (ASYNCcnt tracked).
__device__ __forceinline__ void async_copy16(_Float16* lds_dst, const _Float16* gsrc) {
  unsigned lo = (unsigned)(uintptr_t)lds_dst;            // LDS offset = low 32 bits
  unsigned long long ga = (unsigned long long)(uintptr_t)gsrc;
  asm volatile("global_load_async_to_lds_b128 %0, %1, off"
               :: "v"(lo), "v"(ga) : "memory");
}
__device__ __forceinline__ void wait_async0() {
  asm volatile("s_wait_asynccnt 0x0" ::: "memory");
}

// ---------------- prologue: fp32 -> fp16, fusing the z concat ----------------
__global__ __launch_bounds__(256)
void convert_fp16(const float* __restrict__ xx, const float* __restrict__ pt,
                  const float* __restrict__ pl,
                  const float* __restrict__ Wi, const float* __restrict__ Wf,
                  const float* __restrict__ Wo, const float* __restrict__ Ws,
                  _Float16* __restrict__ Zh, _Float16* __restrict__ Wh) {
  long c = (long)blockIdx.x * blockDim.x + threadIdx.x;
  const float* src;
  _Float16* dst;
  if (c < ZCH) {
    int m = (int)(c / (INDIM / 8));
    int k = (int)(c % (INDIM / 8)) * 8;
    if (k < XD)             src = xx + (size_t)m * XD   + k;
    else if (k < XD + HDIM) src = pt + (size_t)m * HDIM + (k - XD);
    else                    src = pl + (size_t)m * HDIM + (k - XD - HDIM);
    dst = Zh + (size_t)m * INDIM + k;
  } else {
    long c2 = c - ZCH;
    int n = (int)(c2 / (INDIM / 8));            // 0..4095 = gate*1024 + h
    int k = (int)(c2 % (INDIM / 8)) * 8;
    int gate = n >> 10, h = n & 1023;
    const float* Wg = (gate == 0) ? Wi : (gate == 1) ? Wf : (gate == 2) ? Wo : Ws;
    src = Wg + (size_t)h * INDIM + k;
    dst = Wh + (size_t)n * INDIM + k;
  }
  float4 a = ((const float4*)src)[0];
  float4 b = ((const float4*)src)[1];
  v8h o;
  o[0] = (_Float16)a.x; o[1] = (_Float16)a.y; o[2] = (_Float16)a.z; o[3] = (_Float16)a.w;
  o[4] = (_Float16)b.x; o[5] = (_Float16)b.y; o[6] = (_Float16)b.z; o[7] = (_Float16)b.w;
  *(v8h*)dst = o;
}

// ---------------- GEMM helpers ----------------
__device__ __forceinline__ void issue_slab(_Float16* zbuf, _Float16* wbuf,
    const _Float16* zg0, const _Float16* zg1,
    const _Float16* wg0, const _Float16* wg1,
    int d0, int d1, int koff) {
  async_copy16(zbuf + d0, zg0 + koff);
  async_copy16(zbuf + d1, zg1 + koff);
  async_copy16(wbuf + d0, wg0 + koff);
  async_copy16(wbuf + d1, wg1 + koff);
}

__device__ __forceinline__ void compute_slab(const _Float16* zb, const _Float16* wb,
    int wm, int wh, int r, int s, v8f acc[4][2]) {
  // A fragments (16-bit A 16x32 layout): lanes 0-15 K0..7/K16..23, lanes 16-31 K8..15/K24..31
  V16 a[2];
  #pragma unroll
  for (int mt = 0; mt < 2; ++mt) {
    const _Float16* p = zb + (wm * 32 + mt * 16 + r) * LDSS;
    a[mt].h[0] = *(const v8h*)(p + s * 8);
    a[mt].h[1] = *(const v8h*)(p + 16 + s * 8);
  }
  // B 32x16 per gate: lanes 0-15 hold K0..7/K16..23 of column n, lanes 16-31 K8..15/K24..31
  #pragma unroll
  for (int g = 0; g < 4; ++g) {
    const _Float16* q = wb + (g * BH + wh * 16 + r) * LDSS + s * 8;
    V16 b;
    b.h[0] = *(const v8h*)(q);
    b.h[1] = *(const v8h*)(q + 16);
    #pragma unroll
    for (int mt = 0; mt < 2; ++mt) {
      acc[g][mt] = __builtin_amdgcn_wmma_f32_16x16x32_f16(
          false, a[mt].v, false, b.v, (short)0, acc[g][mt], false, false);
    }
  }
}

// ---------------- main GEMM: async LDS staging + fp16 WMMA + fused LSTM ----------------
__global__ __launch_bounds__(256)
void lstm_wmma_async(const _Float16* __restrict__ Zh, const _Float16* __restrict__ Wh,
                     const float* __restrict__ old_state,
                     const float* __restrict__ bi, const float* __restrict__ bfv,
                     const float* __restrict__ bo, const float* __restrict__ bsv,
                     float* __restrict__ out) {
  __shared__ __align__(16) _Float16 ldsZ0[BM * LDSS];
  __shared__ __align__(16) _Float16 ldsZ1[BM * LDSS];
  __shared__ __align__(16) _Float16 ldsW0[4 * BH * LDSS];
  __shared__ __align__(16) _Float16 ldsW1[4 * BH * LDSS];

  const int tid  = threadIdx.x;
  const int lane = tid & 31;
  const int wv   = tid >> 5;          // wave 0..7
  const int wm   = wv & 3;            // wave M position -> 32 rows
  const int wh   = wv >> 2;           // wave H position -> 16 cols
  const int r    = lane & 15;
  const int s    = lane >> 4;

  const int m0 = blockIdx.x * BM;
  const int h0 = blockIdx.y * BH;

  // async loader: thread owns 16-byte chunks tid and tid+256 of each tile
  const int c0 = tid, c1 = tid + 256;
  const int zr0 = c0 >> 2, zo0 = (c0 & 3) * 8;
  const int zr1 = c1 >> 2, zo1 = (c1 & 3) * 8;
  const _Float16* zg0 = Zh + (size_t)(m0 + zr0) * INDIM + zo0;
  const _Float16* zg1 = Zh + (size_t)(m0 + zr1) * INDIM + zo1;
  const _Float16* wg0 = Wh + (size_t)((zr0 >> 5) * HDIM + h0 + (zr0 & 31)) * INDIM + zo0;
  const _Float16* wg1 = Wh + (size_t)((zr1 >> 5) * HDIM + h0 + (zr1 & 31)) * INDIM + zo1;
  const int d0 = zr0 * LDSS + zo0;
  const int d1 = zr1 * LDSS + zo1;

  const v8f vzero = {0.f, 0.f, 0.f, 0.f, 0.f, 0.f, 0.f, 0.f};
  v8f acc[4][2];
  #pragma unroll
  for (int g = 0; g < 4; ++g)
    #pragma unroll
    for (int mt = 0; mt < 2; ++mt)
      acc[g][mt] = vzero;

  // prime buffer 0 with slab 0
  issue_slab(ldsZ0, ldsW0, zg0, zg1, wg0, wg1, d0, d1, 0);

  // steady state: manually double-buffered, branch-free body.
  // Iteration (i, i+1): compute slabs i (buf0) and i+1 (buf1), issue i+1 and i+2.
  #pragma unroll 1
  for (int i = 0; i < NK - 2; i += 2) {
    wait_async0();
    __syncthreads();
    issue_slab(ldsZ1, ldsW1, zg0, zg1, wg0, wg1, d0, d1, (i + 1) * KT);
    compute_slab(ldsZ0, ldsW0, wm, wh, r, s, acc);

    wait_async0();
    __syncthreads();
    issue_slab(ldsZ0, ldsW0, zg0, zg1, wg0, wg1, d0, d1, (i + 2) * KT);
    compute_slab(ldsZ1, ldsW1, wm, wh, r, s, acc);
  }
  // tail: slabs NK-2 (buf0) and NK-1 (buf1)
  wait_async0();
  __syncthreads();
  issue_slab(ldsZ1, ldsW1, zg0, zg1, wg0, wg1, d0, d1, (NK - 1) * KT);
  compute_slab(ldsZ0, ldsW0, wm, wh, r, s, acc);

  wait_async0();
  __syncthreads();
  compute_slab(ldsZ1, ldsW1, wm, wh, r, s, acc);

  // fused LSTM epilogue. C layout: VGPR j -> M = j + 8*s, N = r
  const int hcol = h0 + wh * 16 + r;
  const float b_i = bi[hcol], b_f = bfv[hcol], b_o = bo[hcol], b_s = bsv[hcol];
  #pragma unroll
  for (int mt = 0; mt < 2; ++mt) {
    const int mbase = m0 + wm * 32 + mt * 16 + 8 * s;
    #pragma unroll
    for (int j = 0; j < 8; ++j) {
      const int m = mbase + j;
      float gi = acc[0][mt][j] + b_i;
      float gf = acc[1][mt][j] + b_f;
      float go = acc[2][mt][j] + b_o;
      float gs = acc[3][mt][j] + b_s;
      float os = old_state[(size_t)m * HDIM + hcol];
      float ns = fast_sig(gf) * os + fast_sig(gi) * fast_tanh(gs);
      out[(size_t)m * HDIM + hcol] = fast_sig(go) * fast_tanh(ns);
    }
  }
}

// ---------------- fallback: fused on-the-fly fp16 conversion (no workspace) ----------------
__device__ __forceinline__ const float* z_ptr(const float* xx, const float* pt,
                                              const float* pl, int m, int kg) {
  if (kg < XD)          return xx + (size_t)m * XD   + kg;
  if (kg < XD + HDIM)   return pt + (size_t)m * HDIM + (kg - XD);
  return pl + (size_t)m * HDIM + (kg - XD - HDIM);
}

__global__ __launch_bounds__(256)
void lstm_wmma_fused(const float* __restrict__ xx, const float* __restrict__ pt,
                     const float* __restrict__ pl, const float* __restrict__ old_state,
                     const float* __restrict__ Wi, const float* __restrict__ bi,
                     const float* __restrict__ Wf, const float* __restrict__ bfv,
                     const float* __restrict__ Wo, const float* __restrict__ bo,
                     const float* __restrict__ Ws, const float* __restrict__ bsv,
                     float* __restrict__ out) {
  __shared__ __align__(16) _Float16 zS[BM * LDSS];
  __shared__ __align__(16) _Float16 wS[4 * BH * LDSS];

  const int tid  = threadIdx.x;
  const int lane = tid & 31;
  const int wv   = tid >> 5;
  const int wm   = wv & 3, wh = wv >> 2;
  const int r    = lane & 15, s = lane >> 4;
  const int m0 = blockIdx.x * BM, h0 = blockIdx.y * BH;

  const int zr  = tid >> 1;
  const int kco = (tid & 1) * 16;
  const int gate = zr >> 5, hr = zr & 31;
  const float* Wg   = (gate == 0) ? Wi : (gate == 1) ? Wf : (gate == 2) ? Wo : Ws;
  const float* wsrc = Wg + (size_t)(h0 + hr) * INDIM;

  const v8f vzero = {0.f, 0.f, 0.f, 0.f, 0.f, 0.f, 0.f, 0.f};
  v8f acc[4][2];
  #pragma unroll
  for (int g = 0; g < 4; ++g)
    #pragma unroll
    for (int mt = 0; mt < 2; ++mt)
      acc[g][mt] = vzero;

  #pragma unroll 1
  for (int k0 = 0; k0 < INDIM; k0 += KT) {
    const float* zp = z_ptr(xx, pt, pl, m0 + zr, k0 + kco);
    const float* wp = wsrc + k0 + kco;
    v8h zq0, zq1, wq0, wq1;
    #pragma unroll
    for (int i = 0; i < 2; ++i) {
      float4 a = ((const float4*)zp)[i];
      float4 b = ((const float4*)zp)[i + 2];
      float4 c = ((const float4*)wp)[i];
      float4 d = ((const float4*)wp)[i + 2];
      zq0[4*i+0]=(_Float16)a.x; zq0[4*i+1]=(_Float16)a.y; zq0[4*i+2]=(_Float16)a.z; zq0[4*i+3]=(_Float16)a.w;
      zq1[4*i+0]=(_Float16)b.x; zq1[4*i+1]=(_Float16)b.y; zq1[4*i+2]=(_Float16)b.z; zq1[4*i+3]=(_Float16)b.w;
      wq0[4*i+0]=(_Float16)c.x; wq0[4*i+1]=(_Float16)c.y; wq0[4*i+2]=(_Float16)c.z; wq0[4*i+3]=(_Float16)c.w;
      wq1[4*i+0]=(_Float16)d.x; wq1[4*i+1]=(_Float16)d.y; wq1[4*i+2]=(_Float16)d.z; wq1[4*i+3]=(_Float16)d.w;
    }
    _Float16* zd = &zS[zr * LDSS + kco];
    _Float16* wd = &wS[zr * LDSS + kco];
    *(v8h*)(zd) = zq0; *(v8h*)(zd + 8) = zq1;
    *(v8h*)(wd) = wq0; *(v8h*)(wd + 8) = wq1;
    __syncthreads();
    compute_slab(zS, wS, wm, wh, r, s, acc);
    __syncthreads();
  }

  const int hcol = h0 + wh * 16 + r;
  const float b_i = bi[hcol], b_f = bfv[hcol], b_o = bo[hcol], b_s = bsv[hcol];
  #pragma unroll
  for (int mt = 0; mt < 2; ++mt) {
    const int mbase = m0 + wm * 32 + mt * 16 + 8 * s;
    #pragma unroll
    for (int j = 0; j < 8; ++j) {
      const int m = mbase + j;
      float gi = acc[0][mt][j] + b_i;
      float gf = acc[1][mt][j] + b_f;
      float go = acc[2][mt][j] + b_o;
      float gs = acc[3][mt][j] + b_s;
      float os = old_state[(size_t)m * HDIM + hcol];
      float ns = fast_sig(gf) * os + fast_sig(gi) * fast_tanh(gs);
      out[(size_t)m * HDIM + hcol] = fast_sig(go) * fast_tanh(ns);
    }
  }
}

extern "C" void kernel_launch(void* const* d_in, const int* in_sizes, int n_in,
                              void* d_out, int out_size, void* d_ws, size_t ws_size,
                              hipStream_t stream) {
  const float* xx  = (const float*)d_in[0];
  const float* pt  = (const float*)d_in[1];
  const float* pl  = (const float*)d_in[2];
  const float* os  = (const float*)d_in[3];
  const float* Wi  = (const float*)d_in[4];
  const float* bi  = (const float*)d_in[5];
  const float* Wf  = (const float*)d_in[6];
  const float* bf_ = (const float*)d_in[7];
  const float* Wo  = (const float*)d_in[8];
  const float* bo  = (const float*)d_in[9];
  const float* Ws  = (const float*)d_in[10];
  const float* bs  = (const float*)d_in[11];
  float* out = (float*)d_out;

  const size_t needZ = (size_t)BSZ * INDIM * sizeof(_Float16);      // ~17.8 MB
  const size_t needW = (size_t)4 * HDIM * INDIM * sizeof(_Float16); // ~35.7 MB
  dim3 grid(BSZ / BM, HDIM / BH);                                   // 16 x 32

  if (ws_size >= needZ + needW) {
    _Float16* Zh = (_Float16*)d_ws;
    _Float16* Wh = (_Float16*)((char*)d_ws + needZ);
    convert_fp16<<<dim3((ZCH + WCH) / 256), dim3(256), 0, stream>>>(
        xx, pt, pl, Wi, Wf, Wo, Ws, Zh, Wh);
    lstm_wmma_async<<<grid, dim3(256), 0, stream>>>(
        Zh, Wh, os, bi, bf_, bo, bs, out);
  } else {
    lstm_wmma_fused<<<grid, dim3(256), 0, stream>>>(
        xx, pt, pl, os, Wi, bi, Wf, bf_, Wo, bo, Ws, bs, out);
  }
}
